// GemmaAttention_64458869178908
// MI455X (gfx1250) — matrix operand order repaired and checked by reading, not compiled, not run on hardware
//
#include <hip/hip_runtime.h>
#include <hip/hip_bf16.h>

// ---------------------------------------------------------------------------
// Gemma-style attention for MI455X (gfx1250), bf16 WMMA pipeline:
//   1) Q/K projections      : f32 x f32 -> bf16 ws         (wmma bf16, f32 acc)
//      V projection         : f32 x f32 -> bf16 ws, stored TRANSPOSED (D x B*S)
//   2) RoPE on Q,K (in ws)
//   3) scores = Q K^T/16+M  : bf16 x bf16 -> f32 d_out     (probs region)
//   4) row softmax in place on probs (f32; attn_weights is a returned output)
//   5) attn = P V           : f32(P->bf16 on stage) x bf16 -> bf16 ws
//   6) out = attn Wo^T      : bf16 x f32 -> f32 d_out
// Memory-bound overall (268MB probs dominates). bf16 operands are staged into
// LDS with GLOBAL_LOAD_ASYNC_TO_LDS_B128 (ASYNCcnt) — builtin confirmed present
// by the round-2 diagnostic, which also revealed its v4i-pointer params.
// f32 operands take the load->cvt->ds_store path. Double-buffered LDS with
// issue-before-compute / commit-after-compute ordering (DS ops are in-order per
// wave, so ds_stores must come after the fragment ds_loads of the live buffer).
// ---------------------------------------------------------------------------

typedef __attribute__((ext_vector_type(16))) __bf16       v16bf;
typedef __attribute__((ext_vector_type(8)))  float        v8f;
typedef __attribute__((ext_vector_type(4)))  float        v4f;
typedef __attribute__((ext_vector_type(4)))  __bf16       v4bf;
typedef __attribute__((ext_vector_type(4)))  unsigned int v4u;
typedef __attribute__((ext_vector_type(4)))  int          v4i;

union Frag16 { v16bf bf; v4u u[2]; };

#if defined(__gfx1250__) && __has_builtin(__builtin_amdgcn_global_load_async_to_lds_b128)
#define HAVE_ASYNC_LDS 1
typedef __attribute__((address_space(1))) v4i global_v4i;  // prints as "v4i __device__*"
typedef __attribute__((address_space(3))) v4i lds_v4i;
// generic->AS casts via integer detour (LDS generic addr low 32 bits = LDS offset)
#define TO_AS1(p) ((global_v4i*)(unsigned long long)(p))
#define TO_AS3(p) ((lds_v4i*)(unsigned int)(unsigned long long)(p))
#else
#define HAVE_ASYNC_LDS 0
#endif

template<typename T>
inline constexpr bool use_async_v = (HAVE_ASYNC_LDS != 0) && (sizeof(T) == 2);

__device__ __forceinline__ void wait_async_copies()
{
#if HAVE_ASYNC_LDS
 #if __has_builtin(__builtin_amdgcn_s_wait_asynccnt)
  __builtin_amdgcn_s_wait_asynccnt(0);
 #else
  asm volatile("s_wait_asynccnt 0" ::: "memory");
 #endif
#endif
}

__device__ __forceinline__ v4bf load_quad(const float* p) {
  v4f f = *(const v4f*)p;
  v4bf r;
  r[0] = (__bf16)f[0]; r[1] = (__bf16)f[1];
  r[2] = (__bf16)f[2]; r[3] = (__bf16)f[3];
  return r;
}
__device__ __forceinline__ v4bf load_quad(const __bf16* p) { return *(const v4bf*)p; }

constexpr int BM = 128, BN = 128, BK = 32;
constexpr int LDT = 40;   // padded LDS row (bf16): 80B stride = 16B-aligned, spreads banks

template<typename T> struct StageRegs { v4bf q[4]; };

// Issue the global side of staging a BMxBK (or BNxBK) tile at (row0, k0).
// bf16 sources: fire async DMA directly into LDS. f32 sources: loads into regs.
template<typename T>
__device__ __forceinline__ void stage_issue(StageRegs<T>& R, const T* __restrict__ src,
                                            int ld, int row0, int k0, int tid,
                                            __bf16* dst)
{
  if constexpr (use_async_v<T>) {
#if HAVE_ASYNC_LDS
    #pragma unroll
    for (int it = 0; it < 2; ++it) {            // 512 x 16B chunks, 2 per thread
      int idx = tid + it * 256;
      int r = idx >> 2, c = (idx & 3) << 3;     // c in bf16 units (16B chunks)
      __builtin_amdgcn_global_load_async_to_lds_b128(
          TO_AS1(src + (size_t)(row0 + r) * ld + k0 + c),
          TO_AS3(dst + r * LDT + c), 0, 0);
    }
#endif
  } else {
    __builtin_prefetch(src + (size_t)(row0 + (tid >> 3)) * ld + k0 + 2 * BK, 0, 0);
    #pragma unroll
    for (int it = 0; it < 4; ++it) {            // 1024 quads, 4 per thread
      int idx = tid + it * 256;
      int r = idx >> 3, c = (idx & 7) << 2;
      R.q[it] = load_quad(src + (size_t)(row0 + r) * ld + k0 + c);
    }
  }
}

// Commit the LDS side (ds_store of converted regs). No-op for async sources.
template<typename T>
__device__ __forceinline__ void stage_commit(StageRegs<T>& R, __bf16* dst, int tid)
{
  if constexpr (!use_async_v<T>) {
    #pragma unroll
    for (int it = 0; it < 4; ++it) {
      int idx = tid + it * 256;
      int r = idx >> 3, c = (idx & 7) << 2;
      *(v4bf*)&dst[r * LDT + c] = R.q[it];
    }
  }
}

// C[M,N] = alpha * A[M,K] x B[N,K]^T (+ mask). CT: store C transposed (C[n][m]).
template<typename TA, typename TB, typename TC, bool CT, bool MASK>
__global__ __launch_bounds__(256)
void gemm_wmma_kernel(const TA* __restrict__ Ag, int lda, long long sAb, long long sAh,
                      const TB* __restrict__ Bg, int ldb, long long sBb, long long sBh,
                      TC* __restrict__ Cg, int ldc, long long sCb, long long sCh,
                      const float* __restrict__ Mg, int ldm, long long sMb,
                      float alpha, int Kdim, int nheads)
{
  __shared__ __align__(16) __bf16 sA[2][BM * LDT];
  __shared__ __align__(16) __bf16 sB[2][BN * LDT];

  const int z = blockIdx.z;
  const int bidx = z / nheads, hidx = z % nheads;
  Ag += (size_t)bidx * sAb + (size_t)hidx * sAh;
  Bg += (size_t)bidx * sBb + (size_t)hidx * sBh;
  Cg += (size_t)bidx * sCb + (size_t)hidx * sCh;
  if constexpr (MASK) Mg += (size_t)bidx * sMb;

  const int m0 = blockIdx.y * BM, n0 = blockIdx.x * BN;
  const int tid  = threadIdx.x;
  const int lane = tid & 31, wave = tid >> 5;
  const int wm = wave >> 1, wn = wave & 1;      // 4x2 wave grid; wave owns 32x64

  v8f acc[2][4];
  #pragma unroll
  for (int i = 0; i < 2; ++i)
    #pragma unroll
    for (int j = 0; j < 4; ++j)
      #pragma unroll
      for (int r = 0; r < 8; ++r) acc[i][j][r] = 0.0f;

  StageRegs<TA> ra;
  StageRegs<TB> rb;

  // prologue: fully stage tile 0 into buffer 0
  stage_issue(ra, Ag, lda, m0, 0, tid, sA[0]);
  stage_issue(rb, Bg, ldb, n0, 0, tid, sB[0]);
  stage_commit(ra, sA[0], tid);
  stage_commit(rb, sB[0], tid);

  int buf = 0;
  for (int k0 = 0; k0 < Kdim; k0 += BK) {
    if constexpr (use_async_v<TA> || use_async_v<TB>) wait_async_copies();
    __syncthreads();                             // buffer `buf` published

    const bool has_next = (k0 + BK < Kdim);
    if (has_next) {                              // issue next tile's global traffic
      stage_issue(ra, Ag, lda, m0, k0 + BK, tid, sA[buf ^ 1]);
      stage_issue(rb, Bg, ldb, n0, k0 + BK, tid, sB[buf ^ 1]);
    }

    // ---- fragment loads per ISA layout (from live buffer) ----
    const __bf16* cA = sA[buf];
    const __bf16* cB = sB[buf];
    Frag16 afr[2], bfr[4];
    const int ar = lane & 15, ak = (lane >> 4) << 3;   // A: K chunks at ak, ak+16
    #pragma unroll
    for (int mt = 0; mt < 2; ++mt) {
      int row = wm * 32 + mt * 16 + ar;
      afr[mt].u[0] = *(const v4u*)&cA[row * LDT + ak];
      afr[mt].u[1] = *(const v4u*)&cA[row * LDT + ak + 16];
    }
    const int bc = lane & 15, bk = (lane >> 4) << 3;   // B: K chunks at bk, bk+16
    #pragma unroll
    for (int nt = 0; nt < 4; ++nt) {
      int col = wn * 64 + nt * 16 + bc;
      bfr[nt].u[0] = *(const v4u*)&cB[col * LDT + bk];
      bfr[nt].u[1] = *(const v4u*)&cB[col * LDT + bk + 16];
    }

    #pragma unroll
    for (int mt = 0; mt < 2; ++mt)
      #pragma unroll
      for (int nt = 0; nt < 4; ++nt)
        acc[mt][nt] = __builtin_amdgcn_wmma_f32_16x16x32_bf16(
            false, afr[mt].bf, false, bfr[nt].bf, (short)0, acc[mt][nt],
            false, false);

    if (has_next) {                              // ds_store AFTER the fragment loads
      stage_commit(ra, sA[buf ^ 1], tid);
      stage_commit(rb, sB[buf ^ 1], tid);
    }
    buf ^= 1;
  }

  // ---- epilogue: C/D layout VGPR r -> row r + (lane>>4)*8, col lane&15 ----
  const int cr = (lane >> 4) << 3;
  const int cc = lane & 15;
  #pragma unroll
  for (int mt = 0; mt < 2; ++mt)
    #pragma unroll
    for (int nt = 0; nt < 4; ++nt) {
      int row0 = m0 + wm * 32 + mt * 16 + cr;
      int col  = n0 + wn * 64 + nt * 16 + cc;
      #pragma unroll
      for (int r = 0; r < 8; ++r) {
        float v = acc[mt][nt][r] * alpha;
        if constexpr (MASK) v += Mg[(size_t)(row0 + r) * ldm + col];
        if constexpr (CT)
          Cg[(size_t)col * ldc + (row0 + r)] = (TC)v;
        else
          Cg[(size_t)(row0 + r) * ldc + col] = (TC)v;
      }
    }
}

// RoPE exactly per reference: rotate_half(x) = concat(-x1, x1), x1 = first half.
//   out[i]     = q[i]*cos - q[i]*sin         (i < 128)
//   out[i+128] = q[i+128]*cos + q[i]*sin
__global__ __launch_bounds__(128)
void rope_kernel(__bf16* __restrict__ Q, __bf16* __restrict__ Kc,
                 const int* __restrict__ pos, int nheads_q)
{
  const int token = blockIdx.x;          // b*S + s
  const int i = threadIdx.x;             // 0..127 (freq index)
  const float p = (float)pos[token];
  const float inv = __expf(-(float)i * (9.210340371976184f / 128.0f)); // 10000^(-i/128)
  float sn, cs;
  __sincosf(p * inv, &sn, &cs);

  size_t base = (size_t)token * (size_t)(nheads_q * 256);
  for (int h = 0; h < nheads_q; ++h) {
    size_t o = base + (size_t)h * 256 + i;
    float lo = (float)Q[o], hi = (float)Q[o + 128];
    Q[o]       = (__bf16)(lo * cs - lo * sn);
    Q[o + 128] = (__bf16)(hi * cs + lo * sn);
  }
  {
    size_t o = (size_t)token * 256 + i;
    float lo = (float)Kc[o], hi = (float)Kc[o + 128];
    Kc[o]       = (__bf16)(lo * cs - lo * sn);
    Kc[o + 128] = (__bf16)(hi * cs + lo * sn);
  }
}

// One 256-thread workgroup per row of 2048 f32; in-place softmax.
__global__ __launch_bounds__(256)
void softmax_kernel(float* __restrict__ P, int ncols)
{
  __shared__ float red[256];
  float* row = P + (size_t)blockIdx.x * ncols;
  const int t = threadIdx.x;

  v4f a = *(const v4f*)(row + t * 8);
  v4f b = *(const v4f*)(row + t * 8 + 4);
  float m = fmaxf(fmaxf(fmaxf(a[0], a[1]), fmaxf(a[2], a[3])),
                  fmaxf(fmaxf(b[0], b[1]), fmaxf(b[2], b[3])));
  red[t] = m; __syncthreads();
  for (int s = 128; s > 0; s >>= 1) { if (t < s) red[t] = fmaxf(red[t], red[t + s]); __syncthreads(); }
  m = red[0]; __syncthreads();

  float sum = 0.0f;
  #pragma unroll
  for (int i = 0; i < 4; ++i) {
    a[i] = __expf(a[i] - m); sum += a[i];
    b[i] = __expf(b[i] - m); sum += b[i];
  }
  red[t] = sum; __syncthreads();
  for (int s = 128; s > 0; s >>= 1) { if (t < s) red[t] += red[t + s]; __syncthreads(); }
  float inv = 1.0f / red[0];

  *(v4f*)(row + t * 8)     = a * inv;
  *(v4f*)(row + t * 8 + 4) = b * inv;
}

extern "C" void kernel_launch(void* const* d_in, const int* in_sizes, int n_in,
                              void* d_out, int out_size, void* d_ws, size_t ws_size,
                              hipStream_t stream)
{
  (void)in_sizes; (void)n_in; (void)out_size; (void)ws_size;
  constexpr int  Bb = 2, S = 2048, Hh = 2048, NH = 8, D = 256;
  constexpr long long MS = (long long)Bb * S;        // 4096 token rows

  const float* hidden = (const float*)d_in[0];
  const float* mask   = (const float*)d_in[1];
  const int*   pos    = (const int*)d_in[2];
  const float* Wq     = (const float*)d_in[3];
  const float* Wk     = (const float*)d_in[4];
  const float* Wv     = (const float*)d_in[5];
  const float* Wo     = (const float*)d_in[6];

  float* out   = (float*)d_out;                       // (B,S,H) f32
  float* attnP = out + (size_t)Bb * S * Hh;           // (B,NH,S,S) f32 probs

  // workspace (bf16): Q(4096x2048) | K(4096x256) | Vt(256 x 4096) | attn(4096x2048)
  __bf16* Qw = (__bf16*)d_ws;
  __bf16* Kw = Qw + (size_t)MS * Hh;
  __bf16* Vt = Kw + (size_t)MS * D;                   // transposed: Vt[d][b*S+s]
  __bf16* Aw = Vt + (size_t)MS * D;

  dim3 blk(256);

  // 1) Q/K projections (f32 x f32 -> bf16); V projection with transposed store
  gemm_wmma_kernel<float, float, __bf16, false, false>
      <<<dim3(Hh / BN, MS / BM, 1), blk, 0, stream>>>(
      hidden, Hh, 0, 0, Wq, Hh, 0, 0, Qw, Hh, 0, 0, nullptr, 0, 0, 1.0f, Hh, 1);
  gemm_wmma_kernel<float, float, __bf16, false, false>
      <<<dim3(D / BN, MS / BM, 1), blk, 0, stream>>>(
      hidden, Hh, 0, 0, Wk, Hh, 0, 0, Kw, D, 0, 0, nullptr, 0, 0, 1.0f, Hh, 1);
  gemm_wmma_kernel<float, float, __bf16, true, false>     // CT: Vt[d][token]
      <<<dim3(D / BN, MS / BM, 1), blk, 0, stream>>>(
      hidden, Hh, 0, 0, Wv, Hh, 0, 0, Vt, (int)MS, 0, 0, nullptr, 0, 0, 1.0f, Hh, 1);

  // 2) RoPE on Q and K
  rope_kernel<<<dim3((unsigned)MS), dim3(128), 0, stream>>>(Qw, Kw, pos, NH);

  // 3) scores = Q K^T / 16 + mask -> probs region (batched over b,h; async bf16 staging)
  gemm_wmma_kernel<__bf16, __bf16, float, false, true>
      <<<dim3(S / BN, S / BM, Bb * NH), blk, 0, stream>>>(
      Qw, Hh, (long long)S * Hh, (long long)D,
      Kw, D,  (long long)S * D,  0,
      attnP, S, (long long)NH * S * S, (long long)S * S,
      mask, S, (long long)S * S,
      0.0625f, D, NH);

  // 4) softmax in place (attn_weights is a returned output)
  softmax_kernel<<<dim3(Bb * NH * S), dim3(256), 0, stream>>>(attnP, S);

  // 5) attn = P V : A = probs f32 (cvt on stage), B = Vt (D x MS) bf16 (async)
  gemm_wmma_kernel<float, __bf16, __bf16, false, false>
      <<<dim3(D / BN, S / BM, Bb * NH), blk, 0, stream>>>(
      attnP, S, (long long)NH * S * S, (long long)S * S,
      Vt, (int)MS, (long long)S, 0,
      Aw, Hh, (long long)S * Hh, (long long)D,
      nullptr, 0, 0, 1.0f, S, NH);

  // 6) out = attn Wo^T (bf16 async x f32 -> f32)
  gemm_wmma_kernel<__bf16, float, float, false, false>
      <<<dim3(Hh / BN, MS / BM, 1), blk, 0, stream>>>(
      Aw, Hh, 0, 0, Wo, Hh, 0, 0, out, Hh, 0, 0, nullptr, 0, 0, 1.0f, Hh, 1);
}
